// BDHResonatorLayer_82094004896070
// MI455X (gfx1250) — matrix operand order repaired and checked by reading, not compiled, not run on hardware
//
#include <hip/hip_runtime.h>
#include <hip/hip_bf16.h>
#include <math.h>

// Problem dims (fixed by reference)
#define B_  2
#define T_  1024
#define D_  1024
#define H_  16
#define DH_ 64
#define NF_ 256
#define R_  4096
#define M_  (B_*T_)   // 2048 rows for all GEMMs

typedef __attribute__((ext_vector_type(16))) __bf16        v16bf;
typedef __attribute__((ext_vector_type(8)))  float         v8f;
typedef __attribute__((ext_vector_type(8)))  unsigned int  v8u;
typedef __attribute__((__vector_size__(4 * sizeof(int)))) int v4i_t;

// ---- CDNA5 async global->LDS staging (guarded; falls back to b128 copies)
#if defined(__has_builtin)
#  if __has_builtin(__builtin_amdgcn_global_load_async_to_lds_b128) && \
      __has_builtin(__builtin_amdgcn_s_wait_asynccnt)
#    define USE_ASYNC_LDS 1
#  endif
#endif
#ifndef USE_ASYNC_LDS
#  define USE_ASYNC_LDS 0
#endif

__device__ __forceinline__ void copy16_g2l(const unsigned short* g, unsigned short* l){
#if USE_ASYNC_LDS
  __builtin_amdgcn_global_load_async_to_lds_b128(
      (__attribute__((address_space(1))) v4i_t*)(void*)g,
      (__attribute__((address_space(3))) v4i_t*)(void*)l, 0, 0);
#else
  *(uint4*)l = *(const uint4*)g;
#endif
}
__device__ __forceinline__ void staging_wait(){
#if USE_ASYNC_LDS
  __builtin_amdgcn_s_wait_asynccnt(0);
#endif
}

__device__ __forceinline__ unsigned short f2bf(float f){
  unsigned int u = __float_as_uint(f);
  u += 0x7FFFu + ((u >> 16) & 1u);          // round-to-nearest-even
  return (unsigned short)(u >> 16);
}
__device__ __forceinline__ float sigm(float x){ return 1.0f/(1.0f+__expf(-x)); }

// ---------------------------------------------------------------- cvt f32->bf16
__global__ void cvt_f32_bf16(const float* __restrict__ src,
                             unsigned short* __restrict__ dst, int n){
  int i = blockIdx.x*blockDim.x + threadIdx.x;
  if (i < n) dst[i] = f2bf(src[i]);
}

// ---------------------------------------------------------------- cvt + transpose
// src: f32 [K,N] row-major  ->  dst: bf16 [N,K] row-major. Coalesced both sides.
__global__ __launch_bounds__(256) void cvt_transpose_bf16(
    const float* __restrict__ src, unsigned short* __restrict__ dst, int K, int N){
  __shared__ unsigned short tile[32][33];
  const int tx = threadIdx.x;               // 0..31
  const int ty = threadIdx.y;               // 0..7
  const int n0 = blockIdx.x * 32;
  const int k0 = blockIdx.y * 32;
  #pragma unroll
  for (int r = 0; r < 4; ++r){
    const int kk = ty + r*8;
    tile[kk][tx] = f2bf(src[(size_t)(k0 + kk) * N + (n0 + tx)]);
  }
  __syncthreads();
  #pragma unroll
  for (int r = 0; r < 4; ++r){
    const int nn = ty + r*8;
    dst[(size_t)(n0 + nn) * K + (k0 + tx)] = tile[tx][nn];
  }
}

// ---------------------------------------------------------------- WMMA GEMM
// C[M,N] = act(A[M,K] @ W[K,N] + bias). A: bf16 [M,K]; Bt: bf16 [N,K] (pre-
// transposed weights). Block tile 128(M) x 128(N), K-step 32, 8 waves (4x2),
// each wave 2x4 tiles of 16x16 -> 8 x v_wmma_f32_16x16x32_bf16 per K-iter.
// Double-buffered LDS; staging via global_load_async_to_lds_b128 overlapped
// with WMMA compute on the other buffer.
template<int ACT, int OBF>  // ACT: 0 none, 1 relu; OBF: 0 f32 out, 1 bf16 out
__global__ __launch_bounds__(256) void gemm_bf16(
    const unsigned short* __restrict__ A, const unsigned short* __restrict__ Bt,
    const float* __restrict__ bias, void* __restrict__ Cout,
    int M, int N, int K)
{
  __shared__ __align__(16) unsigned short As[2][128*32];  // [m][k] per buffer
  __shared__ __align__(16) unsigned short Bs[2][128*32];  // [n][k] per buffer

  const int tid  = threadIdx.x;
  const int lane = tid & 31;
  const int wid  = tid >> 5;
  const int wm   = wid >> 1;              // 0..3  (32 rows each)
  const int wn   = wid & 1;               // 0..1  (64 cols each)
  const int mblk = blockIdx.y * 128;
  const int nblk = blockIdx.x * 128;
  const int ml   = lane & 15;
  const int hi   = lane >> 4;

  v8f acc[2][4];
  #pragma unroll
  for (int i=0;i<2;i++)
    #pragma unroll
    for (int j=0;j<4;j++)
      #pragma unroll
      for (int r=0;r<8;r++) acc[i][j][r] = 0.0f;

  // stage one 128x32 bf16 tile (2 x 16B segments per thread per operand)
  auto stage = [&](int buf, int k0){
    #pragma unroll
    for (int i=0;i<2;i++){
      const int idx = i*256 + tid;
      const int row = idx >> 2, seg = (idx & 3) << 3;
      copy16_g2l(&A[(size_t)(mblk + row) * K + (k0 + seg)], &As[buf][row*32 + seg]);
      copy16_g2l(&Bt[(size_t)(nblk + row) * K + (k0 + seg)], &Bs[buf][row*32 + seg]);
    }
  };

  const int kIters = K >> 5;
  // prologue: fill buffer 0
  stage(0, 0);
  staging_wait();
  __syncthreads();

  for (int kt = 0; kt < kIters; ++kt){
    const int cur = kt & 1;
    if (kt + 1 < kIters) stage(cur ^ 1, (kt + 1) << 5);     // async into other buf
    if (kt + 2 < kIters)
      __builtin_prefetch(&Bt[(size_t)(nblk + (tid>>1)) * K + ((kt+2)<<5) + ((tid&1)<<4)], 0, 0);

    // Fragments per ISA 7.12.2 (wave32, 16-bit):
    // A: lane -> M=lane&15; VGPR v holds K = (v>>2)*16 + (lane>>4)*8 + (v&3)*2 + j
    // B: lane -> N=lane&15; VGPR v holds K = (v>>2)*16 + (lane>>4)*8 + (v&3)*2 + j
    v16bf afrag[2], bfrag[4];
    #pragma unroll
    for (int im=0; im<2; ++im){
      const int mrow = wm*32 + im*16 + ml;
      v8u tmp;
      #pragma unroll
      for (int v=0; v<8; ++v){
        int kk = ((v>>2)<<4) + (hi<<3) + ((v&3)<<1);     // even -> aligned b32
        tmp[v] = *(const unsigned int*)&As[cur][mrow*32 + kk];
      }
      afrag[im] = __builtin_bit_cast(v16bf, tmp);
    }
    #pragma unroll
    for (int jn=0; jn<4; ++jn){
      const int ncol = wn*64 + jn*16 + ml;
      v8u tmp;
      #pragma unroll
      for (int v=0; v<8; ++v){
        int kk = ((v>>2)<<4) + (hi<<3) + ((v&3)<<1);
        tmp[v] = *(const unsigned int*)&Bs[cur][ncol*32 + kk];
      }
      bfrag[jn] = __builtin_bit_cast(v16bf, tmp);
    }
    #pragma unroll
    for (int im=0; im<2; ++im)
      #pragma unroll
      for (int jn=0; jn<4; ++jn)
        acc[im][jn] = __builtin_amdgcn_wmma_f32_16x16x32_bf16(
            false, afrag[im], false, bfrag[jn],
            (short)0, acc[im][jn], false, false);

    staging_wait();       // this wave's async copies for buf^1 have landed
    __syncthreads();      // all waves done reading cur / writing cur^1
  }

  // epilogue: C/D layout -> lane n = lane&15, VGPR r -> M = r + 8*(lane>>4)
  #pragma unroll
  for (int im=0; im<2; ++im){
    #pragma unroll
    for (int jn=0; jn<4; ++jn){
      const int col = nblk + wn*64 + jn*16 + ml;
      const float bv = bias ? bias[col] : 0.0f;
      #pragma unroll
      for (int r=0; r<8; ++r){
        const int row = mblk + wm*32 + im*16 + hi*8 + r;
        float val = acc[im][jn][r] + bv;
        if (ACT == 1) val = fmaxf(val, 0.0f);
        if (OBF) ((unsigned short*)Cout)[(size_t)row * N + col] = f2bf(val);
        else     ((float*)Cout)[(size_t)row * N + col] = val;
      }
    }
  }
}

// ---------------------------------------------------------------- resonator prep
// in-place: amp_pre -> u_re, phi_pre -> u_im
__global__ void resonator_prep(float* __restrict__ ap, float* __restrict__ pp, int n){
  int i = blockIdx.x*blockDim.x + threadIdx.x;
  if (i >= n) return;
  float a = ap[i], p = pp[i];
  float amp = (a > 20.0f) ? a : log1pf(__expf(a));         // softplus
  float phi = 3.14159265358979323846f * tanhf(p);
  float s, c; __sincosf(phi, &s, &c);
  ap[i] = amp * c;
  pp[i] = amp * s;
}

// ---------------------------------------------------------------- resonator scan
// 512 sequential chains (b,f); writes feat[B,T,3NF] bf16 = [re, im, abs]
__global__ void resonator_scan(const float* __restrict__ ure, const float* __restrict__ uim,
                               const float* __restrict__ omega, const float* __restrict__ retl,
                               const float* __restrict__ theta,
                               unsigned short* __restrict__ feat){
  int tid = blockIdx.x*blockDim.x + threadIdx.x;
  if (tid >= B_*NF_) return;
  const int f = tid & (NF_-1);
  const int b = tid >> 8;                       // NF_==256
  const float dec = sigm(retl[f]);
  float s, c; __sincosf(omega[f], &s, &c);
  const float ar = dec * c, ai = dec * s;
  const float th = theta[f];
  float xr = 0.0f, xi = 0.0f;
  for (int t = 0; t < T_; ++t){
    const size_t ui = ((size_t)(b*T_ + t))*NF_ + f;
    const float pr = ar*xr - ai*xi + ure[ui];
    const float pi = ar*xi + ai*xr + uim[ui];
    const float mag = sqrtf(pr*pr + pi*pi);
    const float g = sigm(mag - th);
    xr = pr*g; xi = pi*g;
    const size_t fb = ((size_t)(b*T_ + t))*(3*NF_);
    feat[fb + f]         = f2bf(xr);
    feat[fb + NF_ + f]   = f2bf(xi);
    feat[fb + 2*NF_ + f] = f2bf(mag*g);          // |Xn| = g*|pre|
  }
}

// ---------------------------------------------------------------- linear attention
// one workgroup per (b,h); thread e owns state column S[:,e]
__global__ __launch_bounds__(64) void linear_attn(
    const float* __restrict__ q, const float* __restrict__ k,
    const float* __restrict__ v, float* __restrict__ y){
  const int bh = blockIdx.x;
  const int b = bh >> 4, h = bh & 15;
  const int e = threadIdx.x;
  __shared__ float qs[64], ks[64], vs[64], ksum[64];
  float S[64];
  #pragma unroll
  for (int d=0; d<64; ++d) S[d] = 0.0f;
  ksum[e] = 0.0f;
  for (int t = 0; t < T_; ++t){
    const size_t base = ((size_t)(b*T_ + t))*D_ + h*DH_;
    __syncthreads();
    qs[e] = q[base+e]; ks[e] = k[base+e]; vs[e] = v[base+e];
    __syncthreads();
    ksum[e] += ks[e];
    __syncthreads();
    const float ve = vs[e];
    float num = 0.0f, den = 0.0f;
    #pragma unroll
    for (int d=0; d<64; ++d){
      S[d] = __builtin_fmaf(ks[d], ve, S[d]);
      num  = __builtin_fmaf(qs[d], S[d], num);
      den  = __builtin_fmaf(qs[d], ksum[d], den);
    }
    y[base+e] = num / (den + 1e-6f);
  }
}

// ---------------------------------------------------------------- per-head LN
// reads f32 attention output, writes bf16 (input to FFN1 GEMM)
__global__ __launch_bounds__(64) void head_ln(const float* __restrict__ y,
                                              const float* __restrict__ w,
                                              const float* __restrict__ bia,
                                              unsigned short* __restrict__ ybf){
  const int g = blockIdx.x;                     // (b*T+t)*H + h
  const int e = threadIdx.x;
  const int h = g & (H_-1);
  const size_t base = (size_t)g * DH_;
  float val = y[base+e];
  __shared__ float s1[64], s2[64];
  s1[e] = val; s2[e] = val*val;
  __syncthreads();
  for (int off = 32; off > 0; off >>= 1){
    if (e < off){ s1[e] += s1[e+off]; s2[e] += s2[e+off]; }
    __syncthreads();
  }
  const float mu  = s1[0] * (1.0f/DH_);
  const float var = s2[0] * (1.0f/DH_) - mu*mu;
  const float rs  = rsqrtf(var + 1e-5f);
  ybf[base+e] = f2bf((val - mu)*rs*w[h*DH_+e] + bia[h*DH_+e]);
}

// ---------------------------------------------------------------- residual + final LN
__global__ __launch_bounds__(256) void final_ln(
    const float* __restrict__ x, const float* __restrict__ ffn,
    const float* __restrict__ w, const float* __restrict__ bia,
    float* __restrict__ out){
  const int row = blockIdx.x, tid = threadIdx.x;
  const size_t base = (size_t)row * D_;
  __shared__ float s1[256], s2[256];
  float z[4]; float sum = 0.0f, sq = 0.0f;
  #pragma unroll
  for (int i=0;i<4;i++){
    const int c = tid + i*256;
    const float val = x[base+c] + ffn[base+c];
    z[i] = val; sum += val; sq += val*val;
  }
  s1[tid] = sum; s2[tid] = sq;
  __syncthreads();
  for (int off = 128; off > 0; off >>= 1){
    if (tid < off){ s1[tid] += s1[tid+off]; s2[tid] += s2[tid+off]; }
    __syncthreads();
  }
  const float mu  = s1[0] * (1.0f/D_);
  const float var = s2[0] * (1.0f/D_) - mu*mu;
  const float rs  = rsqrtf(var + 1e-5f);
  #pragma unroll
  for (int i=0;i<4;i++){
    const int c = tid + i*256;
    out[base+c] = (z[i]-mu)*rs*w[c] + bia[c];
  }
}

// ---------------------------------------------------------------- launcher
extern "C" void kernel_launch(void* const* d_in, const int* in_sizes, int n_in,
                              void* d_out, int out_size, void* d_ws, size_t ws_size,
                              hipStream_t stream){
  (void)in_sizes; (void)n_in; (void)out_size; (void)ws_size;
  const float* x      = (const float*)d_in[0];
  const float* W_amp  = (const float*)d_in[1];
  const float* b_amp  = (const float*)d_in[2];
  const float* W_phi  = (const float*)d_in[3];
  const float* b_phi  = (const float*)d_in[4];
  const float* omega  = (const float*)d_in[5];
  const float* retl   = (const float*)d_in[6];
  const float* theta  = (const float*)d_in[7];
  const float* W_feat = (const float*)d_in[8];
  const float* b_feat = (const float*)d_in[9];
  const float* Wq     = (const float*)d_in[10];
  const float* Wk     = (const float*)d_in[11];
  const float* Wv     = (const float*)d_in[12];
  const float* lnh_w  = (const float*)d_in[13];
  const float* lnh_b  = (const float*)d_in[14];
  const float* W1     = (const float*)d_in[15];
  const float* b1     = (const float*)d_in[16];
  const float* W2     = (const float*)d_in[17];
  const float* b2     = (const float*)d_in[18];
  const float* ln_w   = (const float*)d_in[19];
  const float* ln_b   = (const float*)d_in[20];
  float* out = (float*)d_out;

  // ---- workspace arena
  char* p = (char*)d_ws;
  auto carve = [&](size_t bytes)->char* {
    char* r = p; p += (bytes + 255) & ~(size_t)255; return r;
  };
  unsigned short* wtAmp  = (unsigned short*)carve((size_t)D_*NF_*2);   // [NF,D]
  unsigned short* wtPhi  = (unsigned short*)carve((size_t)D_*NF_*2);   // [NF,D]
  unsigned short* wtFeat = (unsigned short*)carve((size_t)3*NF_*D_*2); // [D,3NF]
  unsigned short* wtQ    = (unsigned short*)carve((size_t)D_*D_*2);
  unsigned short* wtK    = (unsigned short*)carve((size_t)D_*D_*2);
  unsigned short* wtV    = (unsigned short*)carve((size_t)D_*D_*2);
  unsigned short* wt1    = (unsigned short*)carve((size_t)D_*R_*2);    // [R,D]
  unsigned short* wt2    = (unsigned short*)carve((size_t)R_*D_*2);    // [D,R]
  unsigned short* xbf    = (unsigned short*)carve((size_t)M_*D_*2);
  float* ampPre = (float*)carve((size_t)M_*NF_*4);       // -> u_re
  float* phiPre = (float*)carve((size_t)M_*NF_*4);       // -> u_im
  unsigned short* featbf = (unsigned short*)carve((size_t)M_*3*NF_*2);
  unsigned short* hbf    = (unsigned short*)carve((size_t)M_*D_*2);
  float* qbuf   = (float*)carve((size_t)M_*D_*4);
  float* kbuf   = (float*)carve((size_t)M_*D_*4);
  float* vbuf   = (float*)carve((size_t)M_*D_*4);
  float* ybuf   = (float*)carve((size_t)M_*D_*4);
  unsigned short* ybf    = (unsigned short*)carve((size_t)M_*D_*2);
  unsigned short* midbf  = (unsigned short*)carve((size_t)M_*R_*2);
  float* ffn    = (float*)carve((size_t)M_*D_*4);

  // ---- weight convert + transpose: [K,N] f32 -> [N,K] bf16
  dim3 tb(32, 8);
  auto cvtT = [&](const float* s, unsigned short* d, int K, int N){
    dim3 g(N/32, K/32);
    cvt_transpose_bf16<<<g, tb, 0, stream>>>(s, d, K, N);
  };
  cvtT(W_amp,  wtAmp,  D_,     NF_);
  cvtT(W_phi,  wtPhi,  D_,     NF_);
  cvtT(W_feat, wtFeat, 3*NF_,  D_);
  cvtT(Wq,     wtQ,    D_,     D_);
  cvtT(Wk,     wtK,    D_,     D_);
  cvtT(Wv,     wtV,    D_,     D_);
  cvtT(W1,     wt1,    D_,     R_);
  cvtT(W2,     wt2,    R_,     D_);
  // activations: x f32 -> bf16
  cvt_f32_bf16<<<(M_*D_ + 255)/256, 256, 0, stream>>>(x, xbf, M_*D_);

  dim3 blk(256);
  // amp/phi projections: [2048 x 256]
  {
    dim3 g(NF_/128, M_/128);
    gemm_bf16<0,0><<<g, blk, 0, stream>>>(xbf, wtAmp, b_amp, ampPre, M_, NF_, D_);
    gemm_bf16<0,0><<<g, blk, 0, stream>>>(xbf, wtPhi, b_phi, phiPre, M_, NF_, D_);
  }
  // polar construction (in place)
  resonator_prep<<<(M_*NF_ + 255)/256, 256, 0, stream>>>(ampPre, phiPre, M_*NF_);
  // sequential resonator recurrence -> bf16 feat
  resonator_scan<<<(B_*NF_ + 255)/256, 256, 0, stream>>>(ampPre, phiPre, omega, retl, theta, featbf);
  // feat_proj: h = relu(feat @ W_feat + b_feat) -> bf16
  {
    dim3 g(D_/128, M_/128);
    gemm_bf16<1,1><<<g, blk, 0, stream>>>(featbf, wtFeat, b_feat, hbf, M_, D_, 3*NF_);
  }
  // q/k/v projections (q,k relu; v linear) -> f32 for the scan
  {
    dim3 g(D_/128, M_/128);
    gemm_bf16<1,0><<<g, blk, 0, stream>>>(hbf, wtQ, nullptr, qbuf, M_, D_, D_);
    gemm_bf16<1,0><<<g, blk, 0, stream>>>(hbf, wtK, nullptr, kbuf, M_, D_, D_);
    gemm_bf16<0,0><<<g, blk, 0, stream>>>(hbf, wtV, nullptr, vbuf, M_, D_, D_);
  }
  // causal linear attention
  linear_attn<<<B_*H_, 64, 0, stream>>>(qbuf, kbuf, vbuf, ybuf);
  // per-head layernorm -> bf16
  head_ln<<<M_*H_, 64, 0, stream>>>(ybuf, lnh_w, lnh_b, ybf);
  // FFN: mid = relu(y @ W1 + b1) -> bf16 ; ffn = relu(mid @ W2 + b2) -> f32
  {
    dim3 g1(R_/128, M_/128);
    gemm_bf16<1,1><<<g1, blk, 0, stream>>>(ybf, wt1, b1, midbf, M_, R_, D_);
    dim3 g2(D_/128, M_/128);
    gemm_bf16<1,0><<<g2, blk, 0, stream>>>(midbf, wt2, b2, ffn, M_, D_, R_);
  }
  // residual + final layernorm
  final_ln<<<M_, 256, 0, stream>>>(x, ffn, ln_w, ln_b, out);
}
